// SliceHeadsSBL_61710090109736
// MI455X (gfx1250) — matrix operand order repaired and checked by reading, not compiled, not run on hardware
//
#include <hip/hip_runtime.h>

typedef __bf16 bf16_t;
typedef bf16_t v16bf __attribute__((ext_vector_type(16)));
typedef bf16_t v8bf  __attribute__((ext_vector_type(8)));
typedef float  v8f   __attribute__((ext_vector_type(8)));

constexpr int cB = 64, cM = 10, cK = 30, cH = 512, cN = 128, cS = 5, cNH = 8, cDH = 64;

// ---------------- helpers ----------------

__device__ inline float wave_sum(float x) {
  #pragma unroll
  for (int o = 16; o > 0; o >>= 1) x += __shfl_xor(x, o, 32);
  return x;
}
__device__ inline float wave_max(float x) {
  #pragma unroll
  for (int o = 16; o > 0; o >>= 1) { float y = __shfl_xor(x, o, 32); x = x > y ? x : y; }
  return x;
}
__device__ inline v8f v8f_zero() {
  v8f z;
  #pragma unroll
  for (int i = 0; i < 8; ++i) z[i] = 0.f;
  return z;
}

union frag_u { v16bf v; v8bf h[2]; };

// A fragment (16x32 bf16) from LDS, row-major, stride lda elems (lda*2 must be mult of 16).
// Element e of v16bf <-> K = e + (e>=8 ? 8 : 0) + half*8  (ISA 16-bit A 16x32 layout).
__device__ inline v16bf load_frag_a_lds(const bf16_t* As, int lda, int mrow) {
  int lane = threadIdx.x & 31, half = lane >> 4, mr = lane & 15;
  const bf16_t* p = As + (mrow + mr) * lda + half * 8;
  frag_u u;
  u.h[0] = *(const v8bf*)p;         // K = half*8 + 0..7
  u.h[1] = *(const v8bf*)(p + 16);  // K = 16 + half*8 + 0..7
  return u.v;
}
// A fragment directly from global bf16 matrix (rowsValid rows, stride elems), K offset kb.
__device__ inline v16bf load_frag_a_g(const bf16_t* A0, long stride, int mrow, int rowsValid, int kb) {
  int lane = threadIdx.x & 31, half = lane >> 4, mr = lane & 15;
  int row = mrow + mr;
  frag_u u;
  if (row < rowsValid) {
    const bf16_t* p = A0 + (long)row * stride + kb + half * 8;
    u.h[0] = *(const v8bf*)p;
    u.h[1] = *(const v8bf*)(p + 16);
  } else {
    #pragma unroll
    for (int e = 0; e < 16; ++e) u.v[e] = (bf16_t)0.f;
  }
  return u.v;
}
// B fragment from swizzled LDS: per n-tile, 32 slots (slot == lane == half*16+n) of 16
// contiguous bf16 holding W[kb + half*8 + e + (e>=8 ? 8 : 0)][tile*16 + n]. Two ds_load_b128 per fragment.
__device__ inline v16bf load_frag_b_swz(const bf16_t* Wswz, int ntile) {
  int lane = threadIdx.x & 31;
  const bf16_t* p = Wswz + (ntile << 9) + (lane << 4);
  frag_u u;
  u.h[0] = *(const v8bf*)p;
  u.h[1] = *(const v8bf*)(p + 8);
  return u.v;
}
__device__ inline v8f wmma_bf16(v16bf a, v16bf b, v8f c) {
  return __builtin_amdgcn_wmma_f32_16x16x32_bf16(false, a, false, b, (short)0, c, false, false);
}
// Async global->LDS 16B copy (gfx1250, ASYNCcnt-tracked). lds_addr = LDS byte address
// (generic pointer low 32 bits), gp = per-lane global address.
__device__ inline void async_copy_b128(unsigned lds_addr, const void* gp) {
  asm volatile("global_load_async_to_lds_b128 %0, %1, off"
               :: "v"(lds_addr), "v"(gp) : "memory");
}
__device__ inline void wait_async0() {
  asm volatile("s_wait_asynccnt 0x0" ::: "memory");
}

// ---------------- K0: fp32 -> bf16 (vectorized, n multiple of 8) ----------------
__global__ void cvt_bf16(const float* __restrict__ in, bf16_t* __restrict__ out, long n) {
  long t = (long)blockIdx.x * blockDim.x + threadIdx.x;
  long stride = (long)gridDim.x * blockDim.x;
  for (long i = t * 8; i < n; i += stride * 8) {
    v8bf o;
    #pragma unroll
    for (int j = 0; j < 8; ++j) o[j] = (bf16_t)in[i + j];
    *(v8bf*)(out + i) = o;
  }
}

// ---------------- K1: v_shared[s,h] = W_trans[s,h,:].w_shared ; pb[s] ----------------
__global__ void prep_vshared(const float* __restrict__ Wtr, const float* __restrict__ btr,
                             const float* __restrict__ wsh, const float* __restrict__ bsh,
                             float* __restrict__ vsh, float* __restrict__ pb) {
  int t = blockIdx.x * blockDim.x + threadIdx.x;
  if (t < cS * cH) {
    int s = t / cH, h = t % cH;
    const float* wrow = Wtr + ((long)s * cH + h) * cH;
    float acc = 0.f;
    for (int g = 0; g < cH; ++g) acc += wrow[g] * wsh[g];
    vsh[t] = acc;
  } else if (t < cS * cH + cS) {
    int s = t - cS * cH;
    float acc = bsh[0];
    const float* br = btr + (long)s * cH;
    for (int g = 0; g < cH; ++g) acc += br[g] * wsh[g];
    pb[s] = acc;
  }
}

// ---------------- K2: pred_out[b,m,k,s] (folded reps . w_shared) ----------------
__global__ void pred_kernel(const float* __restrict__ ctx, const float* __restrict__ vsh,
                            const float* __restrict__ pb, float* __restrict__ pred) {
  int wave = (blockIdx.x * blockDim.x + threadIdx.x) >> 5;
  int lane = threadIdx.x & 31;
  if (wave >= cB * cM * cK) return;
  const float* row = ctx + (long)wave * cH + lane * 16;
  float c[16];
  #pragma unroll
  for (int i = 0; i < 16; ++i) c[i] = row[i];
  for (int s = 0; s < cS; ++s) {
    const float* v = vsh + (long)s * cH + lane * 16;
    float p = 0.f;
    #pragma unroll
    for (int i = 0; i < 16; ++i) p += c[i] * v[i];
    p = wave_sum(p);
    if (lane == 0) pred[(long)wave * cS + s] = p + pb[s];
  }
}

// ---------------- K3: qin[s, b*M+m, h] = sent[b, alias, h] + slice_alias[s,h] ----------------
__global__ void build_qin(const float* __restrict__ sent, const int* __restrict__ aidx,
                          const float* __restrict__ sl_al, bf16_t* __restrict__ qin) {
  long t = (long)blockIdx.x * blockDim.x + threadIdx.x;
  long total = (long)cS * cB * cM * cH;
  if (t >= total) return;
  int h = (int)(t % cH);
  long r = t / cH;
  int bm = (int)(r % (cB * cM));
  int s = (int)(r / (cB * cM));
  int b = bm / cM, m = bm % cM;
  int n = aidx[b * cM + m];
  qin[t] = (bf16_t)(sent[((long)b * cN + n) * cH + h] + sl_al[(long)s * cH + h]);
}

// ---------------- K4: generic bf16 WMMA GEMM: C[s][R][512] = A[s][R][512] @ W[s][512][512] ----------------
template <bool OUT_BF16>
__global__ void gemm512(const bf16_t* __restrict__ A, long aStrideS,
                        const bf16_t* __restrict__ W, void* __restrict__ Cout,
                        long cStrideS, int R) {
  constexpr int LDA = 40;  // 32 + 8 pad (80B rows, 16B aligned)
  __shared__ __attribute__((aligned(16))) bf16_t As[64 * LDA];     // 5,120 B
  __shared__ __attribute__((aligned(16))) bf16_t Wswz[8 * 512];    // 8,192 B (8 n-tiles)
  int s = blockIdx.z;
  int rowBase = blockIdx.x * 64;
  int colBase = blockIdx.y * 128;
  const bf16_t* Ab = A + (long)s * aStrideS;
  const bf16_t* Wb = W + (long)s * cH * cH;
  int tid = threadIdx.x, wv = tid >> 5;
  int mtile = wv & 3, ngrp = wv >> 2;  // 4 M-tiles x 2 N-groups (4 n-tiles each)
  // per-thread A-tile slot (R is always a multiple of 64 for our launches)
  int ar = tid >> 2, ac0 = (tid & 3) << 3;
  unsigned a_lds = (unsigned)(size_t)(void*)(As + ar * LDA + ac0);
  const bf16_t* a_gp = Ab + (long)(rowBase + ar) * cH + ac0;
  v8f acc[4];
  #pragma unroll
  for (int j = 0; j < 4; ++j) acc[j] = v8f_zero();

  for (int kb = 0; kb < cH; kb += 32) {
    // A chunk 64x32: single async global->LDS b128 per thread (ASYNCcnt path)
    async_copy_b128(a_lds, a_gp + kb);
    // W chunk 32x128, swizzled fragment-major
    for (int idx = tid; idx < 32 * 16; idx += 256) {
      int k = idx >> 4, c0 = (idx & 15) << 3;
      v8bf g = *(const v8bf*)(Wb + (long)(kb + k) * cH + colBase + c0);
      int half = (k >> 3) & 1, e = (k & 7) + ((k >> 4) << 3);
      #pragma unroll
      for (int j = 0; j < 8; ++j) {
        int c = c0 + j;
        Wswz[((c >> 4) << 9) + ((((half << 4) + (c & 15)) << 4) + e)] = g[j];
      }
    }
    if (kb + 32 < cH)
      __builtin_prefetch(Wb + (long)(kb + 32) * cH + colBase + ((tid & 15) << 3), 0, 1);
    wait_async0();
    __syncthreads();
    v16bf af = load_frag_a_lds(As, LDA, mtile * 16);
    #pragma unroll
    for (int j = 0; j < 4; ++j) {
      v16bf bf = load_frag_b_swz(Wswz, ngrp * 4 + j);
      acc[j] = wmma_bf16(af, bf, acc[j]);
    }
    __syncthreads();
  }
  int lane = tid & 31, half = lane >> 4, n = lane & 15;
  #pragma unroll
  for (int j = 0; j < 4; ++j) {
    int col = colBase + (ngrp * 4 + j) * 16 + n;
    #pragma unroll
    for (int r = 0; r < 8; ++r) {
      int row = rowBase + mtile * 16 + r + half * 8;
      if (row < R) {
        float v = acc[j][r];
        long o = (long)s * cStrideS + (long)row * cH + col;
        if (OUT_BF16) ((bf16_t*)Cout)[o] = (bf16_t)v;
        else          ((float*)Cout)[o] = v;
      }
    }
  }
}

// ---------------- K5: stage-1 MHA (per (b,s,m) block, wave = head) ----------------
__global__ void attn1_kernel(const float* __restrict__ q1, const bf16_t* __restrict__ k1,
                             const bf16_t* __restrict__ v1, const float* __restrict__ sl_en,
                             bf16_t* __restrict__ q2in) {
  int m = blockIdx.x % cM;
  int s = (blockIdx.x / cM) % cS;
  int b = blockIdx.x / (cM * cS);
  int wv = threadIdx.x >> 5, lane = threadIdx.x & 31;
  __shared__ float qs[cNH][cDH];
  __shared__ float wts[cNH][cN];
  long rowSM = (long)s * cB * cM + b * cM + m;
  long qbase = rowSM * cH + wv * cDH;
  qs[wv][lane] = q1[qbase + lane];
  qs[wv][lane + 32] = q1[qbase + lane + 32];
  __syncthreads();
  long kvbase = ((long)s * cB + b) * cN * cH;
  float sc[4];
  #pragma unroll
  for (int i = 0; i < 4; ++i) {
    int n = lane + i * 32;
    const v8bf* kr = (const v8bf*)(k1 + kvbase + (long)n * cH + wv * cDH);
    float a = 0.f;
    #pragma unroll
    for (int c8 = 0; c8 < 8; ++c8) {
      v8bf kv = kr[c8];
      #pragma unroll
      for (int j = 0; j < 8; ++j) a += qs[wv][c8 * 8 + j] * (float)kv[j];
    }
    sc[i] = a * 0.125f;  // 1/sqrt(64)
  }
  float mx = wave_max(fmaxf(fmaxf(sc[0], sc[1]), fmaxf(sc[2], sc[3])));
  float sum = 0.f;
  #pragma unroll
  for (int i = 0; i < 4; ++i) { sc[i] = __expf(sc[i] - mx); sum += sc[i]; }
  sum = wave_sum(sum);
  float inv = 1.f / sum;
  #pragma unroll
  for (int i = 0; i < 4; ++i) wts[wv][lane + i * 32] = sc[i] * inv;
  __syncthreads();
  #pragma unroll
  for (int i = 0; i < 2; ++i) {
    int d = lane + i * 32;
    const bf16_t* vr = v1 + kvbase + wv * cDH + d;
    float a = 0.f;
    for (int n = 0; n < cN; ++n) a += wts[wv][n] * (float)vr[(long)n * cH];
    int g = wv * cDH + d;
    q2in[rowSM * cH + g] = (bf16_t)(a + sl_en[(long)s * cH + g]);
  }
}

// ---------------- K6: stage-2 fused: k2 GEMM (WMMA) + masked softmax + indicator ----------------
__global__ void stage2_kernel(const bf16_t* __restrict__ ctxb, const float* __restrict__ ctxf,
                              const float* __restrict__ q2, const bf16_t* __restrict__ Wk2b,
                              const float* __restrict__ Wind, const float* __restrict__ bind,
                              const unsigned char* __restrict__ emask, float* __restrict__ ind_out) {
  __shared__ __attribute__((aligned(16))) bf16_t Wswz[32 * 512];  // 32,768 B (32 n-tiles)
  __shared__ bf16_t k2s[cK * cH];                                 // 30,720 B
  __shared__ float  scs[cNH * cK];                                // 960 B
  __shared__ float  r0[8], r1[8];
  int m = blockIdx.x % cM;
  int s = (blockIdx.x / cM) % cS;
  int b = blockIdx.x / (cM * cS);
  int tid = threadIdx.x, wv = tid >> 5, lane = tid & 31;
  int mtile = wv & 1, ngrp = wv >> 1;  // 2 M-tiles x 4 N-groups (128 cols each)
  const bf16_t* A0 = ctxb + ((long)b * cM + m) * cK * cH;
  const bf16_t* Wb = Wk2b + (long)s * cH * cH;
  v8f acc[8];
  #pragma unroll
  for (int j = 0; j < 8; ++j) acc[j] = v8f_zero();

  for (int kb = 0; kb < cH; kb += 32) {
    __syncthreads();
    for (int idx = tid; idx < 32 * 64; idx += 256) {
      int k = idx >> 6, c0 = (idx & 63) << 3;
      v8bf g = *(const v8bf*)(Wb + (long)(kb + k) * cH + c0);
      int half = (k >> 3) & 1, e = (k & 7) + ((k >> 4) << 3);
      #pragma unroll
      for (int j = 0; j < 8; ++j) {
        int c = c0 + j;
        Wswz[((c >> 4) << 9) + ((((half << 4) + (c & 15)) << 4) + e)] = g[j];
      }
    }
    if (kb + 32 < cH)
      __builtin_prefetch(Wb + (long)(kb + 32) * cH + ((tid & 63) << 3), 0, 1);
    __syncthreads();
    v16bf af = load_frag_a_g(A0, cH, mtile * 16, cK, kb);
    #pragma unroll
    for (int j = 0; j < 8; ++j) {
      v16bf bf = load_frag_b_swz(Wswz, ngrp * 8 + j);
      acc[j] = wmma_bf16(af, bf, acc[j]);
    }
  }
  __syncthreads();
  int half = lane >> 4, n = lane & 15;
  #pragma unroll
  for (int j = 0; j < 8; ++j) {
    int col = ngrp * 128 + j * 16 + n;
    #pragma unroll
    for (int r = 0; r < 8; ++r) {
      int row = mtile * 16 + r + half * 8;
      if (row < cK) k2s[row * cH + col] = (bf16_t)acc[j][r];
    }
  }
  __syncthreads();
  // scores sc2[a,k] = q2_head . k2_row_head / 8, masked
  const float* qrow = q2 + ((long)s * cB * cM + b * cM + m) * cH;
  if (tid < cNH * cK) {
    int a = tid / cK, k = tid % cK;
    float sc = -1e9f;
    if (emask[((long)b * cM + m) * cK + k] == 0) {
      const float* qh = qrow + a * cDH;
      const bf16_t* kh = &k2s[k * cH + a * cDH];
      float d = 0.f;
      for (int i = 0; i < cDH; ++i) d += qh[i] * (float)kh[i];
      sc = d * 0.125f;
    }
    scs[a * cK + k] = sc;
  }
  __syncthreads();
  if (tid < cNH) {
    float mx = -1e30f;
    for (int k = 0; k < cK; ++k) mx = fmaxf(mx, scs[tid * cK + k]);
    float sum = 0.f;
    for (int k = 0; k < cK; ++k) { float e = __expf(scs[tid * cK + k] - mx); scs[tid * cK + k] = e; sum += e; }
    float inv = 1.f / sum;
    for (int k = 0; k < cK; ++k) scs[tid * cK + k] *= inv;
  }
  __syncthreads();
  // alias_matrix (values = original fp32 context) folded straight into indicator head
  float p0 = 0.f, p1 = 0.f;
  const float* cb = ctxf + ((long)b * cM + m) * cK * cH;
  for (int g = tid; g < cH; g += 256) {
    int a = g / cDH;
    float ag = 0.f;
    for (int k = 0; k < cK; ++k) ag += scs[a * cK + k] * cb[(long)k * cH + g];
    p0 += ag * Wind[((long)s * cH + g) * 2 + 0];
    p1 += ag * Wind[((long)s * cH + g) * 2 + 1];
  }
  p0 = wave_sum(p0); p1 = wave_sum(p1);
  if (lane == 0) { r0[wv] = p0; r1[wv] = p1; }
  __syncthreads();
  if (tid == 0) {
    float t0 = bind[s * 2 + 0], t1 = bind[s * 2 + 1];
    for (int w = 0; w < 8; ++w) { t0 += r0[w]; t1 += r1[w]; }
    long o = (((long)b * cS + s) * cM + m) * 2;
    ind_out[o] = t0; ind_out[o + 1] = t1;
  }
}

// ---------------- K7: masked log-softmax + top2 margin + slice attention ----------------
__global__ void combine_kernel(const float* __restrict__ pred, const float* __restrict__ ind,
                               const unsigned char* __restrict__ emask, float* __restrict__ attn) {
  int t = blockIdx.x * blockDim.x + threadIdx.x;
  if (t >= cB * cM) return;
  int b = t / cM, m = t % cM;
  float lg[cS];
  for (int s = 0; s < cS; ++s) {
    float mx = -1e30f;
    for (int k = 0; k < cK; ++k) {
      float v = emask[(long)t * cK + k] ? -1e9f : pred[((long)t * cK + k) * cS + s];
      mx = fmaxf(mx, v);
    }
    float sum = 0.f;
    for (int k = 0; k < cK; ++k) {
      float v = emask[(long)t * cK + k] ? -1e9f : pred[((long)t * cK + k) * cS + s];
      sum += __expf(v - mx);
    }
    float lse = __logf(sum) + mx;
    float v0 = -1e30f, v1 = -1e30f;
    for (int k = 0; k < cK; ++k) {
      float v = (emask[(long)t * cK + k] ? -1e9f : pred[((long)t * cK + k) * cS + s]) - lse;
      if (v > v0) { v1 = v0; v0 = v; } else if (v > v1) v1 = v;
    }
    float margin = (v0 - v1) / (v0 + v1);
    float x0 = ind[(((long)b * cS + s) * cM + m) * 2 + 0];
    float x1 = ind[(((long)b * cS + s) * cM + m) * 2 + 1];
    float mi = fmaxf(x0, x1);
    float ip = x1 - (__logf(__expf(x0 - mi) + __expf(x1 - mi)) + mi);
    lg[s] = (ip + margin) * 4.0f;  // / TEMP=0.25
  }
  float mx = -1e30f;
  for (int s = 0; s < cS; ++s) mx = fmaxf(mx, lg[s]);
  float sum = 0.f; float e[cS];
  for (int s = 0; s < cS; ++s) { e[s] = __expf(lg[s] - mx); sum += e[s]; }
  for (int s = 0; s < cS; ++s) attn[(long)t * cS + s] = e[s] / sum;
}

// ---------------- K8: rew = ctx @ (sum_s attn_s W_trans[s]) + b_eff ; final ----------------
__global__ void rew_kernel(const bf16_t* __restrict__ ctxb, const bf16_t* __restrict__ Wtb,
                           const float* __restrict__ btr, const float* __restrict__ attn,
                           const float* __restrict__ wfin, const float* __restrict__ bfin,
                           float* __restrict__ rew_out, float* __restrict__ final_out) {
  __shared__ __attribute__((aligned(16))) bf16_t Wswz[32 * 512];  // 32,768 B
  __shared__ float  beff[cH];
  __shared__ float  fpart[32 * 4];
  __shared__ float  bwdot;
  int bm = blockIdx.x, tid = threadIdx.x, wv = tid >> 5, lane = tid & 31;
  int mtile = wv & 1, ngrp = wv >> 1;
  float at[cS];
  #pragma unroll
  for (int s = 0; s < cS; ++s) at[s] = attn[(long)bm * cS + s];
  for (int g = tid; g < cH; g += 256) {
    float a = 0.f;
    #pragma unroll
    for (int s = 0; s < cS; ++s) a += at[s] * btr[(long)s * cH + g];
    beff[g] = a;
  }
  const bf16_t* A0 = ctxb + (long)bm * cK * cH;
  v8f acc[8];
  #pragma unroll
  for (int j = 0; j < 8; ++j) acc[j] = v8f_zero();

  for (int kb = 0; kb < cH; kb += 32) {
    __syncthreads();
    for (int idx = tid; idx < 32 * 64; idx += 256) {
      int k = idx >> 6, c0 = (idx & 63) << 3;
      long off = (long)(kb + k) * cH + c0;
      v8bf w0 = *(const v8bf*)(Wtb + 0L * cH * cH + off);
      v8bf w1 = *(const v8bf*)(Wtb + 1L * cH * cH + off);
      v8bf w2 = *(const v8bf*)(Wtb + 2L * cH * cH + off);
      v8bf w3 = *(const v8bf*)(Wtb + 3L * cH * cH + off);
      v8bf w4 = *(const v8bf*)(Wtb + 4L * cH * cH + off);
      int half = (k >> 3) & 1, e = (k & 7) + ((k >> 4) << 3);
      #pragma unroll
      for (int j = 0; j < 8; ++j) {
        float w = at[0] * (float)w0[j] + at[1] * (float)w1[j] + at[2] * (float)w2[j] +
                  at[3] * (float)w3[j] + at[4] * (float)w4[j];
        int c = c0 + j;
        Wswz[((c >> 4) << 9) + ((((half << 4) + (c & 15)) << 4) + e)] = (bf16_t)w;
      }
    }
    __syncthreads();
    v16bf af = load_frag_a_g(A0, cH, mtile * 16, cK, kb);
    #pragma unroll
    for (int j = 0; j < 8; ++j) {
      v16bf bf = load_frag_b_swz(Wswz, ngrp * 8 + j);
      acc[j] = wmma_bf16(af, bf, acc[j]);
    }
  }
  __syncthreads();
  if (tid == 0) {
    float a = 0.f;
    for (int g = 0; g < cH; ++g) a += beff[g] * wfin[g];
    bwdot = a;
  }
  int half = lane >> 4, n = lane & 15;
  float rowpart[8];
  #pragma unroll
  for (int r = 0; r < 8; ++r) rowpart[r] = 0.f;
  #pragma unroll
  for (int j = 0; j < 8; ++j) {
    int col = ngrp * 128 + j * 16 + n;
    float wf = wfin[col];
    #pragma unroll
    for (int r = 0; r < 8; ++r) {
      int row = mtile * 16 + r + half * 8;
      float v = acc[j][r] + beff[col];
      if (row < cK) rew_out[((long)bm * cK + row) * cH + col] = v;
      rowpart[r] += acc[j][r] * wf;
    }
  }
  #pragma unroll
  for (int o = 1; o < 16; o <<= 1) {
    #pragma unroll
    for (int r = 0; r < 8; ++r) rowpart[r] += __shfl_xor(rowpart[r], o, 32);
  }
  if (n == 0) {
    #pragma unroll
    for (int r = 0; r < 8; ++r) {
      int row = mtile * 16 + r + half * 8;
      if (row < 32) fpart[row * 4 + ngrp] = rowpart[r];
    }
  }
  __syncthreads();
  if (tid < cK) {
    float v = fpart[tid * 4 + 0] + fpart[tid * 4 + 1] + fpart[tid * 4 + 2] + fpart[tid * 4 + 3];
    final_out[(long)bm * cK + tid] = v + bwdot + bfin[0];
  }
}

// ---------------- host ----------------
extern "C" void kernel_launch(void* const* d_in, const int* in_sizes, int n_in,
                              void* d_out, int out_size, void* d_ws, size_t ws_size,
                              hipStream_t stream) {
  const float* ctx  = (const float*)d_in[0];
  const float* sent = (const float*)d_in[1];
  const int*   aidx = (const int*)d_in[2];
  const unsigned char* mask = (const unsigned char*)d_in[3];
  const float* Wtr  = (const float*)d_in[4];
  const float* btr  = (const float*)d_in[5];
  const float* Wind = (const float*)d_in[6];
  const float* bind = (const float*)d_in[7];
  const float* Wq   = (const float*)d_in[8];
  const float* Wk   = (const float*)d_in[9];
  const float* Wv   = (const float*)d_in[10];
  const float* Wq2  = (const float*)d_in[11];
  const float* Wk2  = (const float*)d_in[12];
  const float* slal = (const float*)d_in[13];
  const float* slen = (const float*)d_in[14];
  const float* wsh  = (const float*)d_in[15];
  const float* bsh  = (const float*)d_in[16];
  const float* wfin = (const float*)d_in[17];
  const float* bfin = (const float*)d_in[18];

  float* out_final = (float*)d_out;                       // [B,M,K]
  float* out_rew   = out_final + (long)cB * cM * cK;      // [B,M,K,H]
  float* out_pred  = out_rew + (long)cB * cM * cK * cH;   // [B,M,K,S]
  float* out_ind   = out_pred + (long)cB * cM * cK * cS;  // [B,S,M,2]

  const long nCTX = (long)cB * cM * cK * cH;   // 9,830,400
  const long nSENT = (long)cB * cN * cH;       // 4,194,304
  const long nSHH = (long)cS * cH * cH;        // 1,310,720
  const long nSBMH = (long)cS * cB * cM * cH;  // 1,638,400
  const long nSBNH = (long)cS * cB * cN * cH;  // 20,971,520

  char* ws = (char*)d_ws;
  size_t off = 0;
  auto take = [&](size_t bytes) -> char* {
    char* p = ws + off;
    off = (off + bytes + 255) & ~(size_t)255;
    return p;
  };
  bf16_t* ctxb  = (bf16_t*)take(nCTX * 2);
  bf16_t* sentb = (bf16_t*)take(nSENT * 2);
  bf16_t* Wtrb  = (bf16_t*)take(nSHH * 2);
  bf16_t* Wqb   = (bf16_t*)take(nSHH * 2);
  bf16_t* Wkb   = (bf16_t*)take(nSHH * 2);
  bf16_t* Wvb   = (bf16_t*)take(nSHH * 2);
  bf16_t* Wq2b  = (bf16_t*)take(nSHH * 2);
  bf16_t* Wk2b  = (bf16_t*)take(nSHH * 2);
  float*  vsh   = (float*)take((size_t)cS * cH * 4);
  float*  pb    = (float*)take((size_t)cS * 4);
  bf16_t* qin   = (bf16_t*)take(nSBMH * 2);
  float*  q1    = (float*)take(nSBMH * 4);
  bf16_t* k1    = (bf16_t*)take(nSBNH * 2);
  bf16_t* v1    = (bf16_t*)take(nSBNH * 2);
  bf16_t* q2in  = (bf16_t*)take(nSBMH * 2);
  float*  q2    = (float*)take(nSBMH * 4);
  float*  attn  = (float*)take((size_t)cB * cM * cS * 4);

  // K0: convert
  cvt_bf16<<<512, 256, 0, stream>>>(ctx, ctxb, nCTX);
  cvt_bf16<<<512, 256, 0, stream>>>(sent, sentb, nSENT);
  cvt_bf16<<<512, 256, 0, stream>>>(Wtr, Wtrb, nSHH);
  cvt_bf16<<<512, 256, 0, stream>>>(Wq, Wqb, nSHH);
  cvt_bf16<<<512, 256, 0, stream>>>(Wk, Wkb, nSHH);
  cvt_bf16<<<512, 256, 0, stream>>>(Wv, Wvb, nSHH);
  cvt_bf16<<<512, 256, 0, stream>>>(Wq2, Wq2b, nSHH);
  cvt_bf16<<<512, 256, 0, stream>>>(Wk2, Wk2b, nSHH);
  // K1
  prep_vshared<<<(cS * cH + cS + 255) / 256, 256, 0, stream>>>(Wtr, btr, wsh, bsh, vsh, pb);
  // K2: pred_out
  pred_kernel<<<(cB * cM * cK + 7) / 8, 256, 0, stream>>>(ctx, vsh, pb, out_pred);
  // K3: qin
  build_qin<<<(int)((nSBMH + 255) / 256), 256, 0, stream>>>(sent, aidx, slal, qin);
  // K4: q1, k1, v1
  gemm512<false><<<dim3(10, 4, cS), 256, 0, stream>>>(qin, (long)cB * cM * cH, Wqb, q1, (long)cB * cM * cH, cB * cM);
  gemm512<true><<<dim3(128, 4, cS), 256, 0, stream>>>(sentb, 0L, Wkb, k1, (long)cB * cN * cH, cB * cN);
  gemm512<true><<<dim3(128, 4, cS), 256, 0, stream>>>(sentb, 0L, Wvb, v1, (long)cB * cN * cH, cB * cN);
  // K5: stage-1 attention -> q2in
  attn1_kernel<<<cB * cS * cM, 256, 0, stream>>>(q1, k1, v1, slen, q2in);
  // K4: q2
  gemm512<false><<<dim3(10, 4, cS), 256, 0, stream>>>(q2in, (long)cB * cM * cH, Wq2b, q2, (long)cB * cM * cH, cB * cM);
  // K6: stage-2 fused -> ind_out
  stage2_kernel<<<cB * cS * cM, 256, 0, stream>>>(ctxb, ctx, q2, Wk2b, Wind, bind, mask, out_ind);
  // K7: slice attention weights
  combine_kernel<<<10, 64, 0, stream>>>(out_pred, out_ind, mask, attn);
  // K8: rew + final
  rew_kernel<<<cB * cM, 256, 0, stream>>>(ctxb, Wtrb, btr, attn, wfin, bfin, out_rew, out_final);
}
